// HierarchicalGNN_60086592471248
// MI455X (gfx1250) — matrix operand order repaired and checked by reading, not compiled, not run on hardware
//
#include <hip/hip_runtime.h>
#include <math.h>

#define BATCH 512
#define SEQ   41
#define NROW  (BATCH * SEQ)      // 20992
#define DIN   1280
#define DOUT  1024
#define HEADS 4
#define HC    256
#define EPS_LN 1e-5f

typedef __attribute__((ext_vector_type(16))) __bf16 v16bf;
typedef __attribute__((ext_vector_type(8)))  float  v8f;

// ---------------------------------------------------------------------------
// CDNA5 async global->LDS copy (ASYNCcnt-tracked), 16 bytes per lane.
// VDST operand = per-lane LDS byte offset; VADDR = 64-bit global address.
// inst offset applies to BOTH LDS and global addresses (ISA 08 §4.4).
// ---------------------------------------------------------------------------
__device__ __forceinline__ void async_b128(unsigned lds_off, const void* gaddr) {
    asm volatile("global_load_async_to_lds_b128 %0, %1, off"
                 :: "v"(lds_off), "v"(gaddr) : "memory");
}
__device__ __forceinline__ void async_b128_x2(unsigned lds_off, const void* gaddr) {
    asm volatile("global_load_async_to_lds_b128 %0, %1, off\n\t"
                 "global_load_async_to_lds_b128 %0, %1, off offset:16"
                 :: "v"(lds_off), "v"(gaddr) : "memory");
}
__device__ __forceinline__ void wait_async0() {
    asm volatile("s_wait_asynccnt 0" ::: "memory");
}
__device__ __forceinline__ unsigned lds_off_of(const void* p) {
    return (unsigned)(size_t)p;   // flat LDS pointer: low 32 bits = LDS offset
}

// ---------------------------------------------------------------------------
// Weight convert + transpose: src f32 [K, Ncol] -> dst bf16 [Ncol, K]
// ---------------------------------------------------------------------------
__global__ __launch_bounds__(256) void wcvt_k(const float* __restrict__ src,
                                              __bf16* __restrict__ dst,
                                              int K, int Ncol) {
    int idx = blockIdx.x * 256 + threadIdx.x;
    if (idx < K * Ncol) {
        int k = idx / Ncol, n = idx % Ncol;
        dst[(size_t)n * K + k] = (__bf16)src[idx];
    }
}

// ---------------------------------------------------------------------------
// Prep: per row n (1280 feats): g = sigmoid(dot(x, att_w)+att_b),
// write x_bf = bf16(x), enh_bf = bf16(x*g)
// ---------------------------------------------------------------------------
__global__ __launch_bounds__(256) void prep_k(const float* __restrict__ x,
                                              const float* __restrict__ att_w,
                                              const float* __restrict__ att_b,
                                              __bf16* __restrict__ x_bf,
                                              __bf16* __restrict__ enh_bf) {
    int n = blockIdx.x, tid = threadIdx.x;
    __shared__ float red[256];
    float xv[5]; float part = 0.f;
    for (int i = 0; i < 5; ++i) {
        int idx = tid + 256 * i;
        xv[i] = x[(size_t)n * DIN + idx];
        part += xv[i] * att_w[idx];
    }
    red[tid] = part; __syncthreads();
    for (int off = 128; off; off >>= 1) {
        if (tid < off) red[tid] += red[tid + off];
        __syncthreads();
    }
    float gv = 1.f / (1.f + expf(-(red[0] + att_b[0])));
    for (int i = 0; i < 5; ++i) {
        int idx = tid + 256 * i;
        x_bf[(size_t)n * DIN + idx]   = (__bf16)xv[i];
        enh_bf[(size_t)n * DIN + idx] = (__bf16)(xv[i] * gv);
    }
}

// ---------------------------------------------------------------------------
// WMMA GEMM: C[M,Ncol] = A[M,K](bf16) x Bt[Ncol,K](bf16)^T  (+bias)(+Cprev)
// Block tile 128x64, 8 waves (4M x 2N), each wave 32x32 via 2x2 wmma tiles.
// Double-buffered LDS, async global->LDS fills overlap WMMA compute.
// M multiple of 128, K multiple of 32. NMASK only for ragged Ncol (=41).
// ---------------------------------------------------------------------------
template <bool NMASK, bool ADD_BIAS, bool RELU, bool ACCUM, bool OUT_BF16>
__global__ __launch_bounds__(256) void gemm_wmma_k(
        const __bf16* __restrict__ A, const __bf16* __restrict__ Bt,
        const float* __restrict__ bias, const float* __restrict__ Cprev,
        float* __restrict__ Cout, __bf16* __restrict__ CoutBf,
        int M, int Ncol, int K) {
    __shared__ __bf16 As[2][128][40];   // 80B row stride -> conflict-free
    __shared__ __bf16 Bs[2][64][40];

    const int tid  = threadIdx.x;
    const int lane = tid & 31;
    const int wave = tid >> 5;
    const int wm = wave & 3;            // 0..3 (M)
    const int wn = wave >> 2;           // 0..1 (N)
    const int m0 = blockIdx.x * 128;
    const int n0 = blockIdx.y * 64;
    const int half = lane >> 4;         // 0/1
    const int lmod = lane & 15;

    // staging roles: A = 32B/thread (128x32), B = 16B/thread (64x32)
    const int rowA = tid >> 1, kcA = (tid & 1) * 16;
    const int rowB = tid >> 2, kcB = (tid & 3) * 8;
    const int nB = n0 + rowB;
    const bool bValid = !NMASK || (nB < Ncol);
    const __bf16* aBase = A  + (size_t)(m0 + rowA) * K + kcA;
    const __bf16* bBase = Bt + (size_t)(bValid ? nB : 0) * K + kcB;

    if (NMASK && !bValid) {             // dead rows: zero once, never touched
        uint4 z = make_uint4(0u, 0u, 0u, 0u);
        *(uint4*)&Bs[0][rowB][kcB] = z;
        *(uint4*)&Bs[1][rowB][kcB] = z;
    }

    v8f acc[2][2];
    for (int i = 0; i < 2; ++i)
        for (int j = 0; j < 2; ++j)
            for (int r = 0; r < 8; ++r) acc[i][j][r] = 0.f;

    // prologue: async-fill stage 0 with k-tile 0
    async_b128_x2(lds_off_of(&As[0][rowA][kcA]), aBase);
    if (bValid) async_b128(lds_off_of(&Bs[0][rowB][kcB]), bBase);
    wait_async0();
    __syncthreads();

    int st = 0;
    for (int k0 = 0; k0 < K; k0 += 32, st ^= 1) {
        const bool hasNext = (k0 + 32) < K;
        if (hasNext) {                  // fill other stage while we compute
            async_b128_x2(lds_off_of(&As[st ^ 1][rowA][kcA]), aBase + k0 + 32);
            if (bValid) async_b128(lds_off_of(&Bs[st ^ 1][rowB][kcB]), bBase + k0 + 32);
        }

        v16bf afrag[2], bfrag[2];
        for (int sm = 0; sm < 2; ++sm) {
            int r = wm * 32 + sm * 16 + lmod;
            union { uint4 u[2]; v16bf v; } cv;
            cv.u[0] = *(const uint4*)&As[st][r][half * 8];
            cv.u[1] = *(const uint4*)&As[st][r][16 + half * 8];
            afrag[sm] = cv.v;
        }
        for (int sn = 0; sn < 2; ++sn) {
            int r = wn * 32 + sn * 16 + lmod;
            union { uint4 u[2]; v16bf v; } cv;
            cv.u[0] = *(const uint4*)&Bs[st][r][half * 8];
            cv.u[1] = *(const uint4*)&Bs[st][r][16 + half * 8];
            bfrag[sn] = cv.v;
        }
        for (int sm = 0; sm < 2; ++sm)
            for (int sn = 0; sn < 2; ++sn)
                acc[sm][sn] = __builtin_amdgcn_wmma_f32_16x16x32_bf16(
                    false, afrag[sm], false, bfrag[sn],
                    (short)0, acc[sm][sn], false, false);

        if (hasNext) {
            wait_async0();              // my fills of next stage landed
            __syncthreads();            // everyone done reading + filling
        }
    }

    // epilogue: VGPR r -> row (lane/16)*8 + r, col = lane%16
    for (int sm = 0; sm < 2; ++sm)
        for (int sn = 0; sn < 2; ++sn)
            for (int r = 0; r < 8; ++r) {
                int row = m0 + wm * 32 + sm * 16 + half * 8 + r;
                int col = n0 + wn * 32 + sn * 16 + lmod;
                if (!NMASK || col < Ncol) {
                    float v = acc[sm][sn][r];
                    if (ADD_BIAS) v += bias[col];
                    if (ACCUM)    v += Cprev[(size_t)row * Ncol + col];
                    if (RELU)     v = fmaxf(v, 0.f);
                    size_t o = (size_t)row * Ncol + col;
                    if (OUT_BF16) CoutBf[o] = (__bf16)v;
                    else          Cout[o] = v;
                }
            }
}

// ---------------------------------------------------------------------------
// GCN aggregation: 3-pt stencil, deg = {3 ends, 5 interior}, dup edges x2
// out = dinv_t*(dinv_t*h_t + 2*dinv_{t-1}*h_{t-1} + 2*dinv_{t+1}*h_{t+1}) + b
// ---------------------------------------------------------------------------
__global__ __launch_bounds__(256) void gcn_agg_k(const float* __restrict__ h,
                                                 const float* __restrict__ b,
                                                 float* __restrict__ out) {
    const float R3 = 0.5773502691896258f, R5 = 0.4472135954999579f;
    size_t idx = (size_t)blockIdx.x * 256 + threadIdx.x;
    int n = (int)(idx >> 10), c = (int)(idx & 1023);
    int s = n % SEQ;
    float dc = (s == 0 || s == SEQ - 1) ? R3 : R5;
    float accv = dc * h[idx];
    if (s > 0)       accv += 2.f * ((s - 1 == 0)       ? R3 : R5) * h[idx - 1024];
    if (s < SEQ - 1) accv += 2.f * ((s + 1 == SEQ - 1) ? R3 : R5) * h[idx + 1024];
    out[idx] = dc * accv + b[c];
}

// ---------------------------------------------------------------------------
// LayerNorm over 1024 per row
// ---------------------------------------------------------------------------
__global__ __launch_bounds__(256) void ln_k(const float* __restrict__ in,
                                            const float* __restrict__ g,
                                            const float* __restrict__ b,
                                            float* __restrict__ out) {
    int n = blockIdx.x, tid = threadIdx.x;
    __shared__ float red[256];
    const float* x = in + (size_t)n * DOUT;
    float v[4]; float s = 0.f;
    for (int i = 0; i < 4; ++i) { v[i] = x[tid + 256 * i]; s += v[i]; }
    red[tid] = s; __syncthreads();
    for (int off = 128; off; off >>= 1) {
        if (tid < off) red[tid] += red[tid + off];
        __syncthreads();
    }
    float mu = red[0] * (1.f / DOUT);
    __syncthreads();
    float vs = 0.f;
    for (int i = 0; i < 4; ++i) { float d = v[i] - mu; vs += d * d; }
    red[tid] = vs; __syncthreads();
    for (int off = 128; off; off >>= 1) {
        if (tid < off) red[tid] += red[tid + off];
        __syncthreads();
    }
    float rstd = rsqrtf(red[0] * (1.f / DOUT) + EPS_LN);
    for (int i = 0; i < 4; ++i) {
        int c = tid + 256 * i;
        out[(size_t)n * DOUT + c] = (v[i] - mu) * rstd * g[c] + b[c];
    }
}

// ---------------------------------------------------------------------------
// GAT attention logits: al_s[n,h] = dot(h[n, h*256:...], a_src[h]); same dst
// ---------------------------------------------------------------------------
__global__ __launch_bounds__(256) void gat_al_k(const float* __restrict__ h,
                                                const float* __restrict__ a_src,
                                                const float* __restrict__ a_dst,
                                                float* __restrict__ als,
                                                float* __restrict__ ald) {
    int n = blockIdx.x, tid = threadIdx.x;
    __shared__ float rs[256], rd[256];
    int i0 = tid * 4;                // head = tid/64 constant over the 4
    const float* hv = h + (size_t)n * DOUT + i0;
    float ps = 0.f, pd = 0.f;
    for (int j = 0; j < 4; ++j) {
        ps += hv[j] * a_src[i0 + j];
        pd += hv[j] * a_dst[i0 + j];
    }
    rs[tid] = ps; rd[tid] = pd; __syncthreads();
    for (int off = 32; off; off >>= 1) {
        if ((tid & 63) < off) { rs[tid] += rs[tid + off]; rd[tid] += rd[tid + off]; }
        __syncthreads();
    }
    if ((tid & 63) == 0) {
        int head = tid >> 6;
        als[(size_t)n * HEADS + head] = rs[tid];
        ald[(size_t)n * HEADS + head] = rd[tid];
    }
}

// ---------------------------------------------------------------------------
// GAT aggregation over region block: weights 2 for s!=t, 1 for self loop
// ---------------------------------------------------------------------------
__global__ __launch_bounds__(256) void gat_agg_k(const float* __restrict__ h,
                                                 const float* __restrict__ als,
                                                 const float* __restrict__ ald,
                                                 const float* __restrict__ gat_b,
                                                 float* __restrict__ out) {
    int bb = blockIdx.x >> 2, r = blockIdx.x & 3;
    const int start = r * 13;
    const int L = (r < 3) ? 13 : 2;
    int tid = threadIdx.x;
    int nbase = bb * SEQ + start;

    __shared__ float hsh[13 * 1024];
    __shared__ float sas[13 * 4], sad[13 * 4];
    for (int idx = tid; idx < L * 1024; idx += 256)
        hsh[idx] = h[(size_t)nbase * DOUT + idx];
    if (tid < L * 4) {
        sas[tid] = als[(size_t)nbase * HEADS + tid];
        sad[tid] = ald[(size_t)nbase * HEADS + tid];
    }
    __syncthreads();

    int head = tid >> 6;
    int c0 = (tid & 63) * 4;
    for (int t = 0; t < L; ++t) {
        float adt = sad[t * 4 + head];
        float m = -1e30f;
        for (int s = 0; s < L; ++s) {
            float z = sas[s * 4 + head] + adt;
            z = (z > 0.f) ? z : 0.2f * z;
            m = fmaxf(m, z);
        }
        float den = 0.f, a0 = 0.f, a1 = 0.f, a2 = 0.f, a3 = 0.f;
        for (int s = 0; s < L; ++s) {
            float z = sas[s * 4 + head] + adt;
            z = (z > 0.f) ? z : 0.2f * z;
            float w = expf(z - m) * ((s == t) ? 1.f : 2.f);
            den += w;
            const float* hv = &hsh[s * 1024 + head * 256 + c0];
            a0 += w * hv[0]; a1 += w * hv[1]; a2 += w * hv[2]; a3 += w * hv[3];
        }
        float inv = 1.f / den;
        int cc = head * 256 + c0;
        float* o = out + (size_t)(nbase + t) * DOUT + cc;
        o[0] = a0 * inv + gat_b[cc + 0];
        o[1] = a1 * inv + gat_b[cc + 1];
        o[2] = a2 * inv + gat_b[cc + 2];
        o[3] = a3 * inv + gat_b[cc + 3];
    }
}

// ---------------------------------------------------------------------------
// Row softmax over 41 (one wave32 per row)
// ---------------------------------------------------------------------------
__global__ __launch_bounds__(256) void softmax41_k(float* __restrict__ l) {
    int row = blockIdx.x * 8 + (threadIdx.x >> 5);
    int lane = threadIdx.x & 31;
    float* p = l + (size_t)row * SEQ;
    bool has2 = (lane + 32) < SEQ;
    float v0 = p[lane];
    float v1 = has2 ? p[lane + 32] : -1e30f;
    float m = fmaxf(v0, v1);
    for (int off = 16; off; off >>= 1) m = fmaxf(m, __shfl_xor(m, off));
    float e0 = expf(v0 - m);
    float e1 = has2 ? expf(v1 - m) : 0.f;
    float s = e0 + e1;
    for (int off = 16; off; off >>= 1) s += __shfl_xor(s, off);
    float inv = 1.f / s;
    p[lane] = e0 * inv;
    if (has2) p[lane + 32] = e1 * inv;
}

// ---------------------------------------------------------------------------
// sage_in = pw @ x per batch, written as bf16 [N, 1280]
// ---------------------------------------------------------------------------
__global__ __launch_bounds__(256) void pwx_k(const float* __restrict__ pw,
                                             const float* __restrict__ x,
                                             __bf16* __restrict__ sin_bf) {
    int bb = blockIdx.x;
    int d = blockIdx.y * 256 + threadIdx.x;
    __shared__ float ps[SEQ * SEQ];
    for (int idx = threadIdx.x; idx < SEQ * SEQ; idx += 256)
        ps[idx] = pw[(size_t)bb * SEQ * SEQ + idx];
    __syncthreads();
    float xr[SEQ];
    for (int j = 0; j < SEQ; ++j)
        xr[j] = x[((size_t)bb * SEQ + j) * DIN + d];
    for (int i = 0; i < SEQ; ++i) {
        float a = 0.f;
        const float* pr = &ps[i * SEQ];
        for (int j = 0; j < SEQ; ++j) a += pr[j] * xr[j];
        sin_bf[((size_t)bb * SEQ + i) * DIN + d] = (__bf16)a;
    }
}

// ---------------------------------------------------------------------------
// SAGE mean-agg over global edges (i<j): exclusive prefix / max(j,1) -> bf16
// ---------------------------------------------------------------------------
__global__ __launch_bounds__(256) void sage_prefix_k(const __bf16* __restrict__ sin_bf,
                                                     __bf16* __restrict__ agg_bf) {
    int bb = blockIdx.x;
    int d = blockIdx.y * 256 + threadIdx.x;
    float run = 0.f;
    for (int j = 0; j < SEQ; ++j) {
        size_t o = ((size_t)bb * SEQ + j) * DIN + d;
        agg_bf[o] = (__bf16)(run / fmaxf((float)j, 1.f));
        run += (float)sin_bf[o];
    }
}

// ---------------------------------------------------------------------------
// Gated combine: gates = softmax([sh,ga,sa] @ gate_w + gate_b); total
// ---------------------------------------------------------------------------
__global__ __launch_bounds__(256) void gate_k(const float* __restrict__ sh,
                                              const float* __restrict__ ga,
                                              const float* __restrict__ sa,
                                              const float* __restrict__ gw,
                                              const float* __restrict__ gb,
                                              float* __restrict__ total) {
    int n = blockIdx.x, tid = threadIdx.x;
    __shared__ float r0[256], r1[256], r2[256], gg[3];
    float c0[4], c1[4], c2[4];
    float p0 = 0.f, p1 = 0.f, p2 = 0.f;
    for (int j = 0; j < 4; ++j) {
        int i = tid + 256 * j;
        c0[j] = sh[(size_t)n * DOUT + i];
        c1[j] = ga[(size_t)n * DOUT + i];
        c2[j] = sa[(size_t)n * DOUT + i];
        const float* g0 = gw + (size_t)i * 3;
        const float* g1 = gw + (size_t)(DOUT + i) * 3;
        const float* g2 = gw + (size_t)(2 * DOUT + i) * 3;
        p0 += c0[j] * g0[0] + c1[j] * g1[0] + c2[j] * g2[0];
        p1 += c0[j] * g0[1] + c1[j] * g1[1] + c2[j] * g2[1];
        p2 += c0[j] * g0[2] + c1[j] * g1[2] + c2[j] * g2[2];
    }
    r0[tid] = p0; r1[tid] = p1; r2[tid] = p2; __syncthreads();
    for (int off = 128; off; off >>= 1) {
        if (tid < off) {
            r0[tid] += r0[tid + off];
            r1[tid] += r1[tid + off];
            r2[tid] += r2[tid + off];
        }
        __syncthreads();
    }
    if (tid == 0) {
        float a = r0[0] + gb[0], b = r1[0] + gb[1], c = r2[0] + gb[2];
        float m = fmaxf(a, fmaxf(b, c));
        float ea = expf(a - m), eb = expf(b - m), ec = expf(c - m);
        float inv = 1.f / (ea + eb + ec);
        gg[0] = ea * inv; gg[1] = eb * inv; gg[2] = ec * inv;
    }
    __syncthreads();
    float g0 = gg[0], g1 = gg[1], g2 = gg[2];
    for (int j = 0; j < 4; ++j) {
        int i = tid + 256 * j;
        total[(size_t)n * DOUT + i] = g0 * c0[j] + g1 * c1[j] + g2 * c2[j];
    }
}

// ---------------------------------------------------------------------------
// Host launch
// ---------------------------------------------------------------------------
extern "C" void kernel_launch(void* const* d_in, const int* in_sizes, int n_in,
                              void* d_out, int out_size, void* d_ws, size_t ws_size,
                              hipStream_t stream) {
    (void)in_sizes; (void)n_in; (void)out_size; (void)ws_size;
    const float* x      = (const float*)d_in[0];
    const float* att_w  = (const float*)d_in[1];
    const float* att_b  = (const float*)d_in[2];
    const float* gcn_w  = (const float*)d_in[3];
    const float* gcn_b  = (const float*)d_in[4];
    const float* sn_g   = (const float*)d_in[5];
    const float* sn_b   = (const float*)d_in[6];
    const float* gat_w  = (const float*)d_in[7];
    const float* gat_as = (const float*)d_in[8];
    const float* gat_ad = (const float*)d_in[9];
    const float* gat_b  = (const float*)d_in[10];
    const float* mn_g   = (const float*)d_in[11];
    const float* mn_b   = (const float*)d_in[12];
    const float* sage_wl = (const float*)d_in[13];
    const float* sage_bl = (const float*)d_in[14];
    const float* sage_wr = (const float*)d_in[15];
    const float* fp_w1  = (const float*)d_in[16];
    const float* fp_b1  = (const float*)d_in[17];
    const float* fp_w2  = (const float*)d_in[18];
    const float* fp_b2  = (const float*)d_in[19];
    const float* dn_g   = (const float*)d_in[20];
    const float* dn_b   = (const float*)d_in[21];
    const float* gate_w = (const float*)d_in[22];
    const float* gate_b = (const float*)d_in[23];

    float* out0 = (float*)d_out;                          // shallow
    float* out1 = out0 + (size_t)NROW * DOUT;             // gat
    float* out2 = out1 + (size_t)NROW * DOUT;             // total

    char* ws = (char*)d_ws;
    size_t off = 0;
    auto alloc = [&](size_t bytes) { size_t o = off; off += (bytes + 255) & ~(size_t)255; return o; };
    __bf16* xbf    = (__bf16*)(ws + alloc((size_t)NROW * DIN * 2));
    __bf16* bufB   = (__bf16*)(ws + alloc((size_t)NROW * DIN * 2));  // enh -> sage_in
    __bf16* bufC   = (__bf16*)(ws + alloc((size_t)NROW * DIN * 2));  // relu -> agg
    __bf16* wt_gcn = (__bf16*)(ws + alloc((size_t)DOUT * DIN * 2));
    __bf16* wt_gat = (__bf16*)(ws + alloc((size_t)DOUT * DIN * 2));
    __bf16* wt_wl  = (__bf16*)(ws + alloc((size_t)DOUT * DIN * 2));
    __bf16* wt_wr  = (__bf16*)(ws + alloc((size_t)DOUT * DIN * 2));
    __bf16* wt_fp1 = (__bf16*)(ws + alloc((size_t)512 * DIN * 2));
    __bf16* wt_fp2 = (__bf16*)(ws + alloc((size_t)SEQ * 512 * 2));
    float*  buf1   = (float*)(ws + alloc((size_t)NROW * DOUT * 4));  // h_gcn -> sage acc
    float*  buf2   = (float*)(ws + alloc((size_t)NROW * DOUT * 4));  // h_gat
    float*  buf3   = (float*)(ws + alloc((size_t)NROW * DOUT * 4));  // tmp (agg/LN)
    float*  logits = (float*)(ws + alloc((size_t)NROW * SEQ * 4));   // -> pw
    float*  als    = (float*)(ws + alloc((size_t)NROW * HEADS * 4));
    float*  ald    = (float*)(ws + alloc((size_t)NROW * HEADS * 4));

    dim3 blk(256);
    auto cdiv = [](int a, int b) { return (a + b - 1) / b; };

    // 1) convert+transpose weights
    wcvt_k<<<cdiv(DIN * DOUT, 256), blk, 0, stream>>>(gcn_w, wt_gcn, DIN, DOUT);
    wcvt_k<<<cdiv(DIN * DOUT, 256), blk, 0, stream>>>(gat_w, wt_gat, DIN, DOUT);
    wcvt_k<<<cdiv(DIN * DOUT, 256), blk, 0, stream>>>(sage_wl, wt_wl, DIN, DOUT);
    wcvt_k<<<cdiv(DIN * DOUT, 256), blk, 0, stream>>>(sage_wr, wt_wr, DIN, DOUT);
    wcvt_k<<<cdiv(DIN * 512, 256), blk, 0, stream>>>(fp_w1, wt_fp1, DIN, 512);
    wcvt_k<<<cdiv(512 * SEQ, 256), blk, 0, stream>>>(fp_w2, wt_fp2, 512, SEQ);

    // 2) gating + bf16 casts
    prep_k<<<NROW, blk, 0, stream>>>(x, att_w, att_b, xbf, bufB);

    dim3 gFull(NROW / 128, DOUT / 64);      // 164 x 16

    // 3-5) shallow branch: GCN GEMM -> stencil agg -> LN
    gemm_wmma_k<false, false, false, false, false><<<gFull, blk, 0, stream>>>(
        bufB, wt_gcn, nullptr, nullptr, buf1, nullptr, NROW, DOUT, DIN);
    gcn_agg_k<<<(NROW * DOUT) / 256, blk, 0, stream>>>(buf1, gcn_b, buf3);
    ln_k<<<NROW, blk, 0, stream>>>(buf3, sn_g, sn_b, out0);

    // 6-9) medium branch: GAT GEMM -> attention logits -> block agg -> LN
    gemm_wmma_k<false, false, false, false, false><<<gFull, blk, 0, stream>>>(
        xbf, wt_gat, nullptr, nullptr, buf2, nullptr, NROW, DOUT, DIN);
    gat_al_k<<<NROW, blk, 0, stream>>>(buf2, gat_as, gat_ad, als, ald);
    gat_agg_k<<<BATCH * 4, blk, 0, stream>>>(buf2, als, ald, gat_b, buf3);
    ln_k<<<NROW, blk, 0, stream>>>(buf3, mn_g, mn_b, out1);

    // 10-14) deep branch: propagation MLP -> softmax -> pw@x -> prefix agg
    dim3 gFp1(NROW / 128, 512 / 64);
    gemm_wmma_k<false, true, true, false, true><<<gFp1, blk, 0, stream>>>(
        xbf, wt_fp1, fp_b1, nullptr, nullptr, bufC, NROW, 512, DIN);
    dim3 gFp2(NROW / 128, 1);
    gemm_wmma_k<true, true, false, false, false><<<gFp2, blk, 0, stream>>>(
        bufC, wt_fp2, fp_b2, nullptr, logits, nullptr, NROW, SEQ, 512);
    softmax41_k<<<NROW / 8, blk, 0, stream>>>(logits);
    dim3 gBD(BATCH, DIN / 256);
    pwx_k<<<gBD, blk, 0, stream>>>(logits, x, bufB);
    sage_prefix_k<<<gBD, blk, 0, stream>>>(bufB, bufC);

    // 15-17) SAGE GEMMs: sage_in@wr then += agg@wl + bias, then LN
    gemm_wmma_k<false, false, false, false, false><<<gFull, blk, 0, stream>>>(
        bufB, wt_wr, nullptr, nullptr, buf1, nullptr, NROW, DOUT, DIN);
    gemm_wmma_k<false, true, false, true, false><<<gFull, blk, 0, stream>>>(
        bufC, wt_wl, sage_bl, buf1, buf1, nullptr, NROW, DOUT, DIN);
    ln_k<<<NROW, blk, 0, stream>>>(buf1, dn_g, dn_b, buf3);

    // 18) gated combine
    gate_k<<<NROW, blk, 0, stream>>>(out0, out1, buf3, gate_w, gate_b, out2);
}
